// Transducer_57054345560829
// MI455X (gfx1250) — compile-verified, matrix operand order checked by reading, not run on hardware
//
#include <hip/hip_runtime.h>
#include <hip/hip_bf16.h>
#include <math.h>

// ---------------------------------------------------------------------------
// RNN-T (Transducer) loss for MI455X (gfx1250, wave32, WMMA, TDM).
// B=8, T=400, U=60 (U1=61), E=D=256, H=512, V=28.
//
//   K1: f = enc@We              (f32 WMMA 16x16x4, LDS-staged A, dbl-buffered B)
//   K2: g = dec@Wd + bf         (same kernel)
//   K3: fused joint: tanh(f+g) @ Wp -> log_softmax -> {blank_lp, lab_lp}
//       (bf16 WMMA 16x16x32; f/g tiles moved by the Tensor Data Mover with
//        hardware LDS padding; Wp pre-packed into B-fragment layout in LDS)
//   K4: anti-diagonal DP wavefront per batch -> loss[b]
// ---------------------------------------------------------------------------

typedef __attribute__((ext_vector_type(2)))  float        v2f;
typedef __attribute__((ext_vector_type(8)))  float        v8f;
typedef __attribute__((ext_vector_type(16))) __bf16       v16bf;
typedef __attribute__((ext_vector_type(4)))  unsigned int u32x4;
typedef __attribute__((ext_vector_type(8)))  int          i32x8;
typedef __attribute__((ext_vector_type(4)))  int          i32x4;

union BF16x16 { unsigned int u[8]; v16bf v; };

__device__ __forceinline__ unsigned int pack_bf16(float a, float b) {
#if __has_builtin(__builtin_amdgcn_perm)
    // one v_perm_b32: result = { b[31:16], a[31:16] }
    return __builtin_amdgcn_perm(__float_as_uint(b), __float_as_uint(a), 0x07060302u);
#else
    return (__float_as_uint(a) >> 16) | (__float_as_uint(b) & 0xFFFF0000u);
#endif
}

#if __has_builtin(__builtin_amdgcn_tanhf)
  #define TANHF(x) __builtin_amdgcn_tanhf(x)
#elif __has_builtin(__builtin_amdgcn_tanh_f32)
  #define TANHF(x) __builtin_amdgcn_tanh_f32(x)
#else
  #define TANHF(x) tanhf(x)
#endif

#if __has_builtin(__builtin_amdgcn_tensor_load_to_lds)
  #define HAVE_TDM 1
#else
  #define HAVE_TDM 0
#endif

__device__ __forceinline__ void wait_tensorcnt0() {
#if __has_builtin(__builtin_amdgcn_s_wait_tensorcnt)
    __builtin_amdgcn_s_wait_tensorcnt((short)0);
#else
    asm volatile("s_wait_tensorcnt 0x0" ::: "memory");
#endif
}

__device__ __forceinline__ unsigned lds_byte_offset(const void* p) {
    // flat LDS aperture: low 32 bits = LDS byte address
    return (unsigned)(unsigned long long)(size_t)p;
}

#if HAVE_TDM
// TDM 2-D f32 tile load, global -> LDS, with hardware padding of 2 dwords
// after every 256 dwords (pad_interval=7, pad_amount code 1) => a 512-dword
// row lands with effective stride 516 dwords (bank-conflict-free).
// tensor_rows < tile_rows rows are zero-filled (OOB reads return 0).
__device__ __forceinline__ void tdm_load_2d_f32(
    const float* gptr, unsigned lds_byte, int tensor_rows, int tile_rows,
    int cols, int row_stride_elems)
{
    unsigned long long ga = (unsigned long long)(size_t)gptr;
    u32x4 g0;
    g0[0] = 1u;                                            // count=1, user D#
    g0[1] = lds_byte;                                      // lds_addr
    g0[2] = (unsigned)(ga & 0xFFFFFFFFu);                  // global_addr lo
    g0[3] = (unsigned)((ga >> 32) & 0x01FFFFFFu) | 0x80000000u;  // hi | type=2
    i32x8 g1;
    g1[0] = (2 << 16) | (1 << 20) | (7 << 22) | (1 << 25); // 4B, pad 2dw/256dw
    g1[1] = (int)((unsigned)cols << 16);                   // tensor_dim0 lo16
    g1[2] = (int)(((unsigned)cols >> 16) |                 // tensor_dim0 hi16
                  ((unsigned)tensor_rows << 16));          // tensor_dim1 lo16
    g1[3] = (int)(((unsigned)tensor_rows >> 16) |          // tensor_dim1 hi16
                  ((unsigned)cols << 16));                 // tile_dim0
    g1[4] = (int)((unsigned)tile_rows);                    // tile_dim1 (dim2=0)
    g1[5] = row_stride_elems;                              // tensor_dim0_stride lo32
    g1[6] = 0;                                             // stride hi / dim1_stride
    g1[7] = 0;
    i32x4 z4 = {0, 0, 0, 0};                               // groups 2/3 unused (2-D)
    i32x8 z8 = {0, 0, 0, 0, 0, 0, 0, 0};
    // 6-arg toolchain form: (g0, g1, g2, g3, extra, cpol)
    __builtin_amdgcn_tensor_load_to_lds(g0, g1, z4, z4, z8, 0);
}
#endif

// ---------------------------------------------------------------------------
// C[MxN] = A[MxK] @ B[KxN] (+ bias[N]), fp32 via V_WMMA_F32_16X16X4_F32.
// Grid: (Mtiles, Ntiles/8). Block: 256 thr = 8 waves, one N-tile per wave,
// shared M-tile of A staged in LDS (coalesced), B global loads double-buffered.
// ---------------------------------------------------------------------------
template<int K, int N>
__global__ __launch_bounds__(256) void gemm_f32_wmma(
    const float* __restrict__ A, const float* __restrict__ Bm,
    float* __restrict__ C, const float* __restrict__ bias, int M)
{
    __shared__ float aS[16][K + 4];

    const int mt   = blockIdx.x;
    const int tid  = threadIdx.x;
    const int wave = tid >> 5;
    const int lane = tid & 31;
    const int half = lane >> 4;
    const int lq   = lane & 15;
    const int nt   = blockIdx.y * 8 + wave;

    for (int i = tid; i < 16 * (K / 4); i += 256) {
        int r  = i / (K / 4);
        int c4 = i % (K / 4);
        int row = mt * 16 + r; if (row >= M) row = M - 1;   // clamp, not branch
        float4 v = ((const float4*)(A + (size_t)row * K))[c4];
        aS[r][c4 * 4 + 0] = v.x; aS[r][c4 * 4 + 1] = v.y;
        aS[r][c4 * 4 + 2] = v.z; aS[r][c4 * 4 + 3] = v.w;
    }
    __syncthreads();

    const int col = nt * 16 + lq;
    const float* bcol = Bm + col;

    v2f bcur;
    bcur.x = bcol[(0 + 2 * half) * N];
    bcur.y = bcol[(1 + 2 * half) * N];

    v8f acc = {};
    #pragma unroll 4
    for (int k0 = 0; k0 < K; k0 += 4) {
        if ((k0 & 31) == 0 && k0 + 32 < K)
            __builtin_prefetch(&bcol[(k0 + 32) * N], 0, 1);   // global_prefetch_b8
        v2f bnext = {0.0f, 0.0f};
        if (k0 + 4 < K) {
            bnext.x = bcol[(k0 + 4 + 2 * half) * N];
            bnext.y = bcol[(k0 + 5 + 2 * half) * N];
        }
        v2f a;
        a.x = aS[lq][k0 + 2 * half];
        a.y = aS[lq][k0 + 2 * half + 1];
        acc = __builtin_amdgcn_wmma_f32_16x16x4_f32(
            false, a, false, bcur, (short)0, acc, false, false);
        bcur = bnext;
    }

    float bv = bias ? bias[col] : 0.0f;
    #pragma unroll
    for (int i = 0; i < 8; ++i) {
        int r = mt * 16 + i + 8 * half;
        if (r < M) C[(size_t)r * N + col] = acc[i] + bv;
    }
}

// ---------------------------------------------------------------------------
// Fused joint + log-softmax. Block = 256 thr (8 waves) per (b, 16 t, 16 u).
// LDS row layout for f/g matches TDM padding: a 512-dword row gets 2 pad
// dwords after each 256 dwords => offset(r,c) = 516*r + c + 2*(c>=256).
// ---------------------------------------------------------------------------
#define HDIM  512
#define LDST  (HDIM + 4)
#define VPAD  32
#define NCHNK (HDIM / 32)

__device__ __forceinline__ int kcol(int c) { return c + ((c >> 8) << 1); }

__global__ __launch_bounds__(256) void joint_kernel(
    const float* __restrict__ f,      // [B*T][H]
    const float* __restrict__ g,      // [B*61 (+pad rows)][H], bf folded in
    const float* __restrict__ Wp,     // [H][V]
    const float* __restrict__ bp,     // [V]
    const int*   __restrict__ tgt,    // [B][U]
    float* __restrict__ blank_lp,     // [B][T][U1]
    float* __restrict__ lab_lp,       // [B][T][U]
    int T, int U1, int V, int t_tiles, int u_tiles)
{
    __shared__ float        fT[16 * LDST];             // 33 KB (reused: logits)
    __shared__ float        gT[16 * LDST];             // 33 KB
    __shared__ unsigned int wpP[NCHNK * 8 * 2 * 32];   // 32 KB, B-fragment order

    const int blk = blockIdx.x;
    const int u_t = blk % u_tiles;
    const int t_t = (blk / u_tiles) % t_tiles;
    const int b   = blk / (u_tiles * t_tiles);
    const int tid = threadIdx.x;
    const int wave = tid >> 5, lane = tid & 31;
    const int half = lane >> 4, lq = lane & 15;

#if HAVE_TDM
    // Wave 0 kicks off both tile DMAs; everyone then packs Wp (overlapped).
    if (wave == 0) {
        tdm_load_2d_f32(f + (size_t)(b * T + t_t * 16) * HDIM,
                        lds_byte_offset(fT), 16, 16, HDIM, HDIM);
        int rv = U1 - u_t * 16; if (rv > 16) rv = 16;   // ragged last u-tile
        tdm_load_2d_f32(g + (size_t)(b * U1 + u_t * 16) * HDIM,
                        lds_byte_offset(gT), rv, 16, HDIM, HDIM);
    }
#else
    for (int i = tid; i < 16 * (HDIM / 4); i += 256) {
        int r = i / (HDIM / 4), c4 = i % (HDIM / 4);
        float4 v = ((const float4*)(f + (size_t)(b * T + t_t * 16 + r) * HDIM))[c4];
        float* dst = &fT[r * LDST + kcol(c4 * 4)];
        dst[0] = v.x; dst[1] = v.y; dst[2] = v.z; dst[3] = v.w;
    }
    for (int i = tid; i < 16 * (HDIM / 4); i += 256) {
        int r = i / (HDIM / 4), c4 = i % (HDIM / 4);
        int u = u_t * 16 + r; if (u > U1 - 1) u = U1 - 1;
        float4 v = ((const float4*)(g + (size_t)(b * U1 + u) * HDIM))[c4];
        float* dst = &gT[r * LDST + kcol(c4 * 4)];
        dst[0] = v.x; dst[1] = v.y; dst[2] = v.z; dst[3] = v.w;
    }
#endif

    // Stage Wp pre-packed in WMMA B-fragment order:
    //   wpP[((c*8 + j)*2 + nt)*32 + lane] = {bf16 Wp[kb][n], bf16 Wp[kb+1][n]}
    for (int i = tid; i < NCHNK * 8 * 2 * 32; i += 256) {
        int ln   = i & 31;
        int nt2  = (i >> 5) & 1;
        int j    = (i >> 6) & 7;
        int c    = i >> 9;
        int h    = ln >> 4, q = ln & 15;
        int kb   = c * 32 + ((j >> 2) << 4) + h * 8 + ((j & 3) << 1);
        int n    = nt2 * 16 + q;
        unsigned int pk = 0u;
        if (n < V)
            pk = pack_bf16(Wp[(size_t)kb * V + n], Wp[(size_t)(kb + 1) * V + n]);
        wpP[i] = pk;
    }
#if HAVE_TDM
    if (wave == 0) wait_tensorcnt0();
#endif
    __syncthreads();

    v8f acc[2][2] = {};                    // [t-subtile][n-tile]
    for (int c = 0; c < NCHNK; ++c) {
        BF16x16 bq[2];
        #pragma unroll
        for (int j = 0; j < 8; ++j) {
            bq[0].u[j] = wpP[(((c << 3) + j) << 6) + lane];
            bq[1].u[j] = wpP[(((c << 3) + j) << 6) + 32 + lane];
        }
        #pragma unroll
        for (int s = 0; s < 2; ++s) {
            const int tl = wave * 2 + s;
            BF16x16 aq;                    // A row = u (lq), fused tanh(f+g)
            #pragma unroll
            for (int j = 0; j < 8; ++j) {
                int kb = c * 32 + ((j >> 2) << 4) + half * 8 + ((j & 3) << 1);
                int kc = kcol(kb);
                float2 fv = *(const float2*)&fT[tl * LDST + kc];   // broadcast
                float2 gv = *(const float2*)&gT[lq * LDST + kc];   // conflict-free
                aq.u[j] = pack_bf16(TANHF(fv.x + gv.x), TANHF(fv.y + gv.y));
            }
            acc[s][0] = __builtin_amdgcn_wmma_f32_16x16x32_bf16(
                false, aq.v, false, bq[0].v, (short)0, acc[s][0], false, false);
            acc[s][1] = __builtin_amdgcn_wmma_f32_16x16x32_bf16(
                false, aq.v, false, bq[1].v, (short)0, acc[s][1], false, false);
        }
    }
    __syncthreads();                       // all waves done reading fT
    float* logits = fT;                    // reuse as [256 rows][32 v]
    #pragma unroll
    for (int s = 0; s < 2; ++s) {
        const int tl = wave * 2 + s;
        #pragma unroll
        for (int nt2 = 0; nt2 < 2; ++nt2)
            #pragma unroll
            for (int i = 0; i < 8; ++i)
                logits[(tl * 16 + (i + 8 * half)) * VPAD + nt2 * 16 + lq] = acc[s][nt2][i];
    }
    __syncthreads();

    // One (t,u) row per thread: log-softmax over 28, emit blank / label lp.
    const int tl = tid >> 4, ul = tid & 15;
    const int t = t_t * 16 + tl;
    const int u = u_t * 16 + ul;
    if (u < U1) {
        float vbuf[28];
        float m = -3.4e38f;
        for (int v = 0; v < V; ++v) {
            float x = logits[tid * VPAD + v] + bp[v];
            vbuf[v] = x; m = fmaxf(m, x);
        }
        float ssum = 0.0f;
        for (int v = 0; v < V; ++v) ssum += __expf(vbuf[v] - m);
        float lse = m + __logf(ssum);
        blank_lp[((size_t)b * T + t) * U1 + u] = vbuf[27] - lse;   // BLANK = 27
        if (u < U1 - 1) {
            int lab = tgt[b * (U1 - 1) + u];
            lab_lp[((size_t)b * T + t) * (U1 - 1) + u] = vbuf[lab] - lse;
        }
    }
}

// ---------------------------------------------------------------------------
// Forward DP, anti-diagonal wavefront. One 64-thread block per batch.
// ---------------------------------------------------------------------------
__global__ __launch_bounds__(64) void rnnt_dp_kernel(
    const float* __restrict__ blank_lp, const float* __restrict__ lab_lp,
    const int* __restrict__ t_lens, const int* __restrict__ u_lens,
    float* __restrict__ loss, int T, int U1)
{
    __shared__ float diag0[64], diag1[64];
    __shared__ float saved;
    const int b = blockIdx.x;
    const int u = threadIdx.x;
    const float* blk = blank_lp + (size_t)b * T * U1;
    const float* lab = lab_lp   + (size_t)b * T * (U1 - 1);
    const int tl = t_lens[b], ulen = u_lens[b];

    if (u == 0) { diag0[0] = 0.0f; saved = 0.0f; }
    __syncthreads();

    float* prev = diag0;
    float* cur  = diag1;
    for (int d = 1; d < T + U1 - 1; ++d) {
        if (u < U1) {
            int lo = d - (T - 1); if (lo < 0) lo = 0;
            int hi = (d < U1 - 1) ? d : (U1 - 1);
            if (u >= lo && u <= hi) {
                int t = d - u;
                float val;
                if (t == 0) {
                    val = prev[u - 1] + lab[u - 1];
                } else if (u == 0) {
                    val = prev[0] + blk[(size_t)(t - 1) * U1];
                } else {
                    float fb = prev[u]     + blk[(size_t)(t - 1) * U1 + u];
                    float fl = prev[u - 1] + lab[(size_t)t * (U1 - 1) + (u - 1)];
                    float m  = fmaxf(fb, fl);
                    val = m + __logf(__expf(fb - m) + __expf(fl - m));
                }
                cur[u] = val;
                if (t == tl - 1 && u == ulen) saved = val;
            }
        }
        __syncthreads();
        float* tmp = prev; prev = cur; cur = tmp;
    }
    if (u == 0)
        loss[b] = -(saved + blk[(size_t)(tl - 1) * U1 + ulen]);
}

// ---------------------------------------------------------------------------
extern "C" void kernel_launch(void* const* d_in, const int* in_sizes, int n_in,
                              void* d_out, int out_size, void* d_ws, size_t ws_size,
                              hipStream_t stream) {
    const float* enc = (const float*)d_in[0];   // [8][400][256]
    const float* dec = (const float*)d_in[1];   // [8][61][256]
    const float* We  = (const float*)d_in[2];   // [256][512]
    const float* Wd  = (const float*)d_in[3];   // [256][512]
    const float* bf  = (const float*)d_in[4];   // [512]
    const float* Wp  = (const float*)d_in[5];   // [512][28]
    const float* bp  = (const float*)d_in[6];   // [28]
    const int*   tgt = (const int*)d_in[7];     // [8][60]
    const int*   ilen= (const int*)d_in[8];     // [8]
    const int*   tlen= (const int*)d_in[9];     // [8]
    float* out = (float*)d_out;

    const int Bb = 8, T = 400, U1 = 61, E = 256, H = 512, V = 28;
    const int Mf = Bb * T;            // 3200
    const int Mg = Bb * U1;           // 488
    const int MgPad = 496;            // 31 tiles of 16

    float* f     = (float*)d_ws;                     // 3200*512
    float* g     = f + (size_t)Mf * H;               // 496*512
    float* blank = g + (size_t)MgPad * H;            // 8*400*61
    float* labp  = blank + (size_t)Bb * T * U1;      // 8*400*60

    // K1: f = enc @ We
    gemm_f32_wmma<256, 512><<<dim3(Mf / 16, (H / 16) / 8), 256, 0, stream>>>(
        enc, We, f, nullptr, Mf);
    // K2: g = dec @ Wd + bf
    gemm_f32_wmma<256, 512><<<dim3(MgPad / 16, (H / 16) / 8), 256, 0, stream>>>(
        dec, Wd, g, bf, Mg);
    // K3: fused joint + log-softmax
    {
        int t_tiles = T / 16;                        // 25
        int u_tiles = (U1 + 15) / 16;                // 4
        joint_kernel<<<Bb * t_tiles * u_tiles, 256, 0, stream>>>(
            f, g, Wp, bp, tgt, blank, labp, T, U1, V, t_tiles, u_tiles);
    }
    // K4: DP wavefront
    rnnt_dp_kernel<<<Bb, 64, 0, stream>>>(blank, labp, ilen, tlen, out, T, U1);
}
